// ChangeDetectorDoubleAttDyn_88021059764525
// MI455X (gfx1250) — compile-verified, matrix operand order checked by reading, not run on hardware
//
#include <hip/hip_runtime.h>

typedef __bf16 bf16;
typedef __attribute__((ext_vector_type(16))) __bf16 v16bf;
typedef __attribute__((ext_vector_type(8)))  float  v8f;

// ---------------- problem constants ----------------
constexpr int BATCH = 64;
constexpr int CIN   = 1024;
constexpr int HH    = 14;
constexpr int WWID  = 14;
constexpr int HW    = 196;
constexpr int DM    = 512;
constexpr int CO    = 256;
constexpr int NCAT  = 768;     // theta|phi|g concatenated
constexpr int HEADS = 8;
constexpr int HD    = 64;
constexpr int MTOT  = BATCH * HW;        // 12544 rows
constexpr long OUTSZ = (long)MTOT * DM;  // floats per output slot

// ---------------- WMMA helpers ----------------
__device__ __forceinline__ v8f vzero8() {
  v8f v;
#pragma unroll
  for (int i = 0; i < 8; ++i) v[i] = 0.0f;
  return v;
}

// A (16x32, row-major source) or B from (N,K)-row-major weights.
__device__ __forceinline__ v16bf frag_rm(const bf16* base, int ld, int r0, int k0, int lane) {
  const int r   = r0 + (lane & 15);
  const int khi = (lane >> 4) & 1;
  const bf16* p0 = base + (size_t)r * ld + k0 + khi * 8;
  v16bf f;
#pragma unroll
  for (int i = 0; i < 8; ++i) { f[i] = p0[i]; f[8 + i] = p0[16 + i]; }
  return f;
}

// B from (K,N)-row-major source (e.g. V matrix: rows = K index, cols = N).
__device__ __forceinline__ v16bf frag_kn(const bf16* base, int ld, int k0, int n0, int lane) {
  const int n   = n0 + (lane & 15);
  const int khi = (lane >> 4) & 1;
  v16bf f;
#pragma unroll
  for (int i = 0; i < 8; ++i) {
    f[i]     = base[(size_t)(k0 + khi * 8 + i) * ld + n];
    f[8 + i] = base[(size_t)(k0 + 16 + khi * 8 + i) * ld + n];
  }
  return f;
}

__device__ __forceinline__ v8f wmma_bf16(v16bf a, v16bf b, v8f c) {
  return __builtin_amdgcn_wmma_f32_16x16x32_bf16(false, a, false, b, (short)0, c, false, false);
}

// ---------------- kernel 0: weight conversion to bf16 ----------------
__global__ void k_cvt(const float* __restrict__ Wimg, const float* __restrict__ Wt,
                      const float* __restrict__ Wp, const float* __restrict__ Wg,
                      const float* __restrict__ Wback,
                      bf16* __restrict__ WimgB, bf16* __restrict__ WcatB,
                      bf16* __restrict__ WbackB) {
  int idx = blockIdx.x * 256 + threadIdx.x;
  const int N1 = DM * CIN;       // 524288
  const int N2 = 2 * NCAT * DM;  // 786432
  const int N3 = 2 * DM * CO;    // 262144
  if (idx < N1) { WimgB[idx] = (bf16)Wimg[idx]; return; }
  idx -= N1;
  if (idx < N2) {
    int l = idx / (NCAT * DM);
    int rem = idx % (NCAT * DM);
    int n = rem / DM, k = rem % DM;
    float v;
    if (n < 256)      v = Wt[(l * 256 + n) * DM + k];
    else if (n < 512) v = Wp[(l * 256 + (n - 256)) * DM + k];
    else              v = Wg[(l * 256 + (n - 512)) * DM + k];
    WcatB[idx] = (bf16)v;
    return;
  }
  idx -= N2;
  if (idx < N3) WbackB[idx] = (bf16)Wback[idx];
}

// ---------------- kernel 1: img_proj = GEMM + bias + LayerNorm + pos-emb ----
// grid(784, 2), block 256 (8 waves). WG computes 16 rows x full 512 cols.
__global__ void k_imgproj(const float* __restrict__ in1, const float* __restrict__ in2,
                          const bf16* __restrict__ Wb, const float* __restrict__ bias,
                          const float* __restrict__ lng, const float* __restrict__ lnb,
                          const float* __restrict__ embw, const float* __restrict__ embh,
                          float* __restrict__ out, bf16* __restrict__ Xb0,
                          bf16* __restrict__ Xb1) {
  const int img = blockIdx.y;
  const float* in = img ? in2 : in1;
  float* X = out + (size_t)img * OUTSZ;
  bf16* Xb = img ? Xb1 : Xb0;
  const int m0 = blockIdx.x * 16;
  const int t = threadIdx.x, lane = t & 31, wave = t >> 5;

  __shared__ __align__(16) bf16 Abuf[16][32];
  __shared__ float Ytile[16][512];
  __shared__ float red1[16][16], red2[16][16];
  __shared__ float rmean[16], rrstd[16];

  // A-staging coords: thread (ar, ak) loads column pair (ak, ak+16)
  const int ar = t & 15;
  const int ak = t >> 4;
  const int am = m0 + ar;
  const int ab = am / HW, ap = am % HW;
  const float* asrc = in + (size_t)ab * CIN * HW + ap;

  v8f acc[4];
#pragma unroll
  for (int i = 0; i < 4; ++i) acc[i] = vzero8();

  for (int k0 = 0; k0 < CIN; k0 += 32) {
    __syncthreads();
    Abuf[ar][ak]      = (bf16)asrc[(size_t)(k0 + ak) * HW];
    Abuf[ar][ak + 16] = (bf16)asrc[(size_t)(k0 + ak + 16) * HW];
    __syncthreads();
    v16bf a = frag_rm(&Abuf[0][0], 32, 0, 0, lane);
#pragma unroll
    for (int tt = 0; tt < 4; ++tt) {
      v16bf b = frag_rm(Wb, CIN, wave * 64 + tt * 16, k0, lane);
      acc[tt] = wmma_bf16(a, b, acc[tt]);
    }
  }
  __syncthreads();
  {
    int col = lane & 15, rb = (lane >> 4) * 8;
#pragma unroll
    for (int tt = 0; tt < 4; ++tt) {
      int n = wave * 64 + tt * 16 + col;
      float bv = bias[n];
#pragma unroll
      for (int i = 0; i < 8; ++i) Ytile[rb + i][n] = acc[tt][i] + bv;
    }
  }
  __syncthreads();
  {
    int r = t >> 4, c = t & 15;
    float s1 = 0.f, s2 = 0.f;
#pragma unroll 4
    for (int i = 0; i < 32; ++i) { float v = Ytile[r][c * 32 + i]; s1 += v; s2 += v * v; }
    red1[r][c] = s1; red2[r][c] = s2;
  }
  __syncthreads();
  if (t < 16) {
    float s1 = 0.f, s2 = 0.f;
    for (int c = 0; c < 16; ++c) { s1 += red1[t][c]; s2 += red2[t][c]; }
    float mean = s1 / 512.0f;
    float var = s2 / 512.0f - mean * mean;
    rmean[t] = mean; rrstd[t] = rsqrtf(var + 1e-6f);
  }
  __syncthreads();
  {
    int r = t >> 4, c = t & 15;
    int m = m0 + r, p = m % HW;
    int ph = p / WWID, pw = p % WWID;
    float mean = rmean[r], rstd = rrstd[r];
    for (int i = 0; i < 32; ++i) {
      int d = c * 32 + i;
      float pe = (d < 256) ? embw[pw * 256 + d] : embh[ph * 256 + (d - 256)];
      float v = (Ytile[r][d] - mean) * rstd * lng[d] + lnb[d] + pe;
      size_t idx = (size_t)m * DM + d;
      X[idx] = v;
      Xb[idx] = (bf16)v;
    }
  }
}

// ---------------- kernel 2: encoder theta|phi|g fused GEMM ----------------
// grid(784), block 256. M=16, N=768, K=512.
__global__ void k_encgemm(const bf16* __restrict__ Xb, const bf16* __restrict__ Wcat,
                          const float* __restrict__ bt, const float* __restrict__ bp,
                          const float* __restrict__ bg, float* __restrict__ TPG) {
  const int m0 = blockIdx.x * 16;
  const int t = threadIdx.x, lane = t & 31, wave = t >> 5;
  const bf16* Arow = Xb + (size_t)m0 * DM;
  v8f acc[6];
#pragma unroll
  for (int i = 0; i < 6; ++i) acc[i] = vzero8();
  for (int k0 = 0; k0 < DM; k0 += 32) {
    v16bf a = frag_rm(Arow, DM, 0, k0, lane);
#pragma unroll
    for (int tt = 0; tt < 6; ++tt) {
      v16bf b = frag_rm(Wcat, DM, wave * 96 + tt * 16, k0, lane);
      acc[tt] = wmma_bf16(a, b, acc[tt]);
    }
  }
  int col = lane & 15, rb = (lane >> 4) * 8;
#pragma unroll
  for (int tt = 0; tt < 6; ++tt) {
    int n = wave * 96 + tt * 16 + col;
    float bv = (n < 256) ? bt[n] : (n < 512 ? bp[n - 256] : bg[n - 512]);
#pragma unroll
    for (int i = 0; i < 8; ++i) TPG[(size_t)(m0 + rb + i) * NCAT + n] = acc[tt][i] + bv;
  }
}

// ---------------- kernel 3: 9-neighbor cosine attention + Wback GEMM -------
// grid(784), block 256.
__global__ void k_encattn(const float* __restrict__ TPG, const bf16* __restrict__ WbackB,
                          const float* __restrict__ bback, float* __restrict__ X,
                          bf16* __restrict__ Xb) {
  const int m0 = blockIdx.x * 16;
  const int t = threadIdx.x, lane = t & 31, wave = t >> 5;
  __shared__ float red[16][16][19];
  __shared__ float smL[16][9];
  __shared__ int   mnL[16][9];
  __shared__ __align__(16) bf16 waL[16][256];

  {  // phase 1: partial dots for theta^2, num_n, phi_n^2
    int r = t >> 4, c = t & 15;
    int m = m0 + r, b = m / HW, p = m % HW, h = p / WWID, w = p % WWID;
    int o0 = c * 16;
    float thv[16];
    float th2 = 0.f;
#pragma unroll
    for (int i = 0; i < 16; ++i) {
      float v = TPG[(size_t)m * NCAT + o0 + i];
      thv[i] = v; th2 += v * v;
    }
    red[r][c][0] = th2;
    for (int n = 0; n < 9; ++n) {
      int di = n / 3 - 1, dj = n % 3 - 1;
      int hh = min(max(h + di, 0), HH - 1);
      int ww = min(max(w + dj, 0), WWID - 1);
      int mn = b * HW + hh * WWID + ww;
      if (c == 0) mnL[r][n] = mn;
      const float* ph = TPG + (size_t)mn * NCAT + 256 + o0;
      float nm = 0.f, p2 = 0.f;
#pragma unroll
      for (int i = 0; i < 16; ++i) { float v = ph[i]; nm += v * thv[i]; p2 += v * v; }
      red[r][c][1 + n] = nm; red[r][c][10 + n] = p2;
    }
  }
  __syncthreads();
  if (t < 16) {  // reduce + cosine softmax over the 9 window positions
    int r = t;
    float s[19];
    for (int q = 0; q < 19; ++q) {
      float a = 0.f;
      for (int c = 0; c < 16; ++c) a += red[r][c][q];
      s[q] = a;
    }
    float nt = fmaxf(sqrtf(s[0]), 1e-8f);
    float cosv[9], mx = -1e30f;
    for (int n = 0; n < 9; ++n) {
      float np = fmaxf(sqrtf(s[10 + n]), 1e-8f);
      cosv[n] = s[1 + n] / (np * nt);
      mx = fmaxf(mx, cosv[n]);
    }
    float ssum = 0.f;
    for (int n = 0; n < 9; ++n) { cosv[n] = __expf(cosv[n] - mx); ssum += cosv[n]; }
    for (int n = 0; n < 9; ++n) smL[r][n] = cosv[n] / ssum;
  }
  __syncthreads();
  {  // phase 2: weighted average of g-projections -> bf16 A operand
    int r = t >> 4, ob = (t & 15) * 16;
    for (int i = 0; i < 16; ++i) {
      int o = ob + i;
      float a = 0.f;
#pragma unroll
      for (int n = 0; n < 9; ++n)
        a += smL[r][n] * TPG[(size_t)mnL[r][n] * NCAT + 512 + o];
      waL[r][o] = (bf16)a;
    }
  }
  __syncthreads();
  // phase 3: out = wa @ Wback^T ; X += out (residual), refresh bf16 shadow
  v8f acc[4];
#pragma unroll
  for (int i = 0; i < 4; ++i) acc[i] = vzero8();
  for (int k0 = 0; k0 < CO; k0 += 32) {
    v16bf a = frag_rm(&waL[0][0], 256, 0, k0, lane);
#pragma unroll
    for (int tt = 0; tt < 4; ++tt) {
      v16bf b = frag_rm(WbackB, CO, wave * 64 + tt * 16, k0, lane);
      acc[tt] = wmma_bf16(a, b, acc[tt]);
    }
  }
  int col = lane & 15, rb = (lane >> 4) * 8;
#pragma unroll
  for (int tt = 0; tt < 4; ++tt) {
    int n = wave * 64 + tt * 16 + col;
    float bv = bback[n];
#pragma unroll
    for (int i = 0; i < 8; ++i) {
      size_t idx = (size_t)(m0 + rb + i) * DM + n;
      float nv = X[idx] + acc[tt][i] + bv;
      X[idx] = nv;
      Xb[idx] = (bf16)nv;
    }
  }
}

// ---------------- kernel 4: cross attention (depthwise conv fused) ---------
// grid(13 qtiles, 64 batch, 16 = head*2dir), block 256.
__global__ void k_cross(const bf16* __restrict__ Xb0, const bf16* __restrict__ Xb1,
                        const float* __restrict__ Wq, const float* __restrict__ Wk,
                        const float* __restrict__ Wv, float* __restrict__ out) {
  const int qt = blockIdx.x;
  const int b = blockIdx.y;
  const int head = blockIdx.z & 7;
  const int dir = blockIdx.z >> 3;
  const bf16* qX = dir ? Xb1 : Xb0;
  const bf16* kX = dir ? Xb0 : Xb1;
  float* ctx = out + (size_t)(2 + dir) * OUTSZ;
  const int t = threadIdx.x, lane = t & 31, wave = t >> 5;
  const int ch0 = head * HD;
  const int row0 = qt * 16;

  __shared__ __align__(16) bf16 kvBuf[224 * 64];  // K conv, reused for V conv
  __shared__ __align__(16) bf16 qBuf[16 * 64];
  __shared__ float sBuf[16 * 208];
  __shared__ __align__(16) bf16 pBuf[16 * 224];
  __shared__ float red1[16][16];
  __shared__ float rowmax[16], rowsum[16];

  // depthwise 3x3 conv (zero pad) for K over all 196 pixels, this head's 64 ch
  for (int idx = t; idx < HW * HD; idx += 256) {
    int p = idx >> 6, o = idx & 63, ch = ch0 + o;
    int h = p / WWID, w = p % WWID;
    float a = 0.f;
#pragma unroll
    for (int ky = 0; ky < 3; ++ky)
#pragma unroll
      for (int kx = 0; kx < 3; ++kx) {
        int hh = h + ky - 1, ww = w + kx - 1;
        if (hh >= 0 && hh < HH && ww >= 0 && ww < WWID)
          a += (float)kX[(size_t)(b * HW + hh * WWID + ww) * DM + ch] * Wk[ch * 9 + ky * 3 + kx];
      }
    kvBuf[p * 64 + o] = (bf16)a;
  }
  for (int idx = t; idx < (208 - HW) * HD; idx += 256) kvBuf[HW * 64 + idx] = (bf16)0.0f;
  // q conv for this 16-row tile
  for (int idx = t; idx < 16 * HD; idx += 256) {
    int rr = idx >> 6, o = idx & 63, ch = ch0 + o;
    int p = row0 + rr;
    float a = 0.f;
    if (p < HW) {
      int h = p / WWID, w = p % WWID;
#pragma unroll
      for (int ky = 0; ky < 3; ++ky)
#pragma unroll
        for (int kx = 0; kx < 3; ++kx) {
          int hh = h + ky - 1, ww = w + kx - 1;
          if (hh >= 0 && hh < HH && ww >= 0 && ww < WWID)
            a += (float)qX[(size_t)(b * HW + hh * WWID + ww) * DM + ch] * Wq[ch * 9 + ky * 3 + kx];
        }
    }
    qBuf[idx] = (bf16)a;
  }
  __syncthreads();

  // scores = q @ k^T / 8 : M=16, N=208 (13 tiles over 8 waves), K=64
  for (int nt = wave; nt < 13; nt += 8) {
    v8f acc = vzero8();
#pragma unroll
    for (int k0 = 0; k0 < HD; k0 += 32) {
      v16bf a = frag_rm(qBuf, 64, 0, k0, lane);
      v16bf bb = frag_rm(kvBuf, 64, nt * 16, k0, lane);
      acc = wmma_bf16(a, bb, acc);
    }
    int col = lane & 15, rb = (lane >> 4) * 8;
    int key = nt * 16 + col;
#pragma unroll
    for (int i = 0; i < 8; ++i) sBuf[(rb + i) * 208 + key] = acc[i] * 0.125f;
  }
  __syncthreads();

  // softmax over keys < 196
  {
    int r = t >> 4, c = t & 15;
    float mx = -1e30f;
    for (int k = c; k < HW; k += 16) mx = fmaxf(mx, sBuf[r * 208 + k]);
    red1[r][c] = mx;
  }
  __syncthreads();
  if (t < 16) {
    float mx = -1e30f;
    for (int c = 0; c < 16; ++c) mx = fmaxf(mx, red1[t][c]);
    rowmax[t] = mx;
  }
  __syncthreads();
  {
    int r = t >> 4, c = t & 15;
    float s = 0.f;
    for (int k = c; k < HW; k += 16) {
      float e = __expf(sBuf[r * 208 + k] - rowmax[r]);
      sBuf[r * 208 + k] = e;
      s += e;
    }
    red1[r][c] = s;
  }
  __syncthreads();
  if (t < 16) {
    float s = 0.f;
    for (int c = 0; c < 16; ++c) s += red1[t][c];
    rowsum[t] = s;
  }
  __syncthreads();
  {
    int r = t >> 4, c = t & 15;
    float inv = 1.0f / rowsum[r];
    for (int k = c; k < 224; k += 16)
      pBuf[r * 224 + k] = (k < HW) ? (bf16)(sBuf[r * 208 + k] * inv) : (bf16)0.0f;
  }
  __syncthreads();

  // depthwise conv for V (reuse kvBuf), zero-pad rows 196..223
  for (int idx = t; idx < HW * HD; idx += 256) {
    int p = idx >> 6, o = idx & 63, ch = ch0 + o;
    int h = p / WWID, w = p % WWID;
    float a = 0.f;
#pragma unroll
    for (int ky = 0; ky < 3; ++ky)
#pragma unroll
      for (int kx = 0; kx < 3; ++kx) {
        int hh = h + ky - 1, ww = w + kx - 1;
        if (hh >= 0 && hh < HH && ww >= 0 && ww < WWID)
          a += (float)kX[(size_t)(b * HW + hh * WWID + ww) * DM + ch] * Wv[ch * 9 + ky * 3 + kx];
      }
    kvBuf[p * 64 + o] = (bf16)a;
  }
  for (int idx = t; idx < (224 - HW) * HD; idx += 256) kvBuf[HW * 64 + idx] = (bf16)0.0f;
  __syncthreads();

  // ctx = probs @ v : M=16, N=64 (4 tiles on waves 0..3), K=224
  if (wave < 4) {
    int nt = wave;
    v8f acc = vzero8();
    for (int k0 = 0; k0 < 224; k0 += 32) {
      v16bf a = frag_rm(pBuf, 224, 0, k0, lane);
      v16bf bb = frag_kn(kvBuf, 64, k0, nt * 16, lane);
      acc = wmma_bf16(a, bb, acc);
    }
    int col = lane & 15, rb = (lane >> 4) * 8;
    int o = ch0 + nt * 16 + col;
#pragma unroll
    for (int i = 0; i < 8; ++i) {
      int p = row0 + rb + i;
      if (p < HW) ctx[(size_t)(b * HW + p) * DM + o] = acc[i];
    }
  }
}

// ---------------- kernel 5: ctx + residual, LayerNorm (in place) -----------
// grid(784, 2), block 256.
__global__ void k_ctxln(float* __restrict__ out, const float* __restrict__ g,
                        const float* __restrict__ bb) {
  const int dir = blockIdx.y;
  const int m0 = blockIdx.x * 16;
  float* ctx = out + (size_t)(2 + dir) * OUTSZ;
  const float* res = out + (size_t)dir * OUTSZ;
  __shared__ float tile[16][512];
  __shared__ float red1[16][16], red2[16][16];
  __shared__ float rmean[16], rrstd[16];
  int t = threadIdx.x, r = t >> 4, c = t & 15;
  float s1 = 0.f, s2 = 0.f;
  for (int i = 0; i < 32; ++i) {
    int d = c * 32 + i;
    size_t idx = (size_t)(m0 + r) * DM + d;
    float v = ctx[idx] + res[idx];
    tile[r][d] = v; s1 += v; s2 += v * v;
  }
  red1[r][c] = s1; red2[r][c] = s2;
  __syncthreads();
  if (t < 16) {
    float a = 0.f, q = 0.f;
    for (int c2 = 0; c2 < 16; ++c2) { a += red1[t][c2]; q += red2[t][c2]; }
    float mean = a / 512.0f;
    float var = q / 512.0f - mean * mean;
    rmean[t] = mean; rrstd[t] = rsqrtf(var + 1e-6f);
  }
  __syncthreads();
  for (int i = 0; i < 32; ++i) {
    int d = c * 32 + i;
    size_t idx = (size_t)(m0 + r) * DM + d;
    ctx[idx] = (tile[r][d] - rmean[r]) * rrstd[r] * g[d] + bb[d];
  }
}

// ---------------- host launcher ----------------
extern "C" void kernel_launch(void* const* d_in, const int* in_sizes, int n_in,
                              void* d_out, int out_size, void* d_ws, size_t ws_size,
                              hipStream_t stream) {
  const float* in1     = (const float*)d_in[0];
  const float* in2     = (const float*)d_in[1];
  const float* W_img   = (const float*)d_in[2];
  const float* b_img   = (const float*)d_in[3];
  const float* ln_g    = (const float*)d_in[4];
  const float* ln_b    = (const float*)d_in[5];
  const float* emb_w   = (const float*)d_in[6];
  const float* emb_h   = (const float*)d_in[7];
  const float* enc_Wt  = (const float*)d_in[8];
  const float* enc_bt  = (const float*)d_in[9];
  const float* enc_Wp  = (const float*)d_in[10];
  const float* enc_bp  = (const float*)d_in[11];
  const float* enc_Wg  = (const float*)d_in[12];
  const float* enc_bg  = (const float*)d_in[13];
  const float* enc_Wbk = (const float*)d_in[14];
  const float* enc_bbk = (const float*)d_in[15];
  const float* ca_Wq   = (const float*)d_in[16];
  const float* ca_Wk   = (const float*)d_in[17];
  const float* ca_Wv   = (const float*)d_in[18];
  const float* ca_ln_g = (const float*)d_in[19];
  const float* ca_ln_b = (const float*)d_in[20];
  float* outp = (float*)d_out;

  // workspace layout
  char* ws = (char*)d_ws;
  size_t off = 0;
  bf16* Xb0 = (bf16*)(ws + off);  off += (size_t)MTOT * DM * sizeof(bf16);
  bf16* Xb1 = (bf16*)(ws + off);  off += (size_t)MTOT * DM * sizeof(bf16);
  float* TPG = (float*)(ws + off); off += (size_t)MTOT * NCAT * sizeof(float);
  bf16* WimgB = (bf16*)(ws + off); off += (size_t)DM * CIN * sizeof(bf16);
  bf16* WcatB = (bf16*)(ws + off); off += (size_t)2 * NCAT * DM * sizeof(bf16);
  bf16* WbackB = (bf16*)(ws + off); off += (size_t)2 * DM * CO * sizeof(bf16);
  (void)ws_size; (void)in_sizes; (void)n_in; (void)out_size;

  // 0) weights -> bf16
  k_cvt<<<6144, 256, 0, stream>>>(W_img, enc_Wt, enc_Wp, enc_Wg, enc_Wbk,
                                  WimgB, WcatB, WbackB);

  // 1) img_proj for both images (writes d_out slots 0/1 + bf16 shadows)
  k_imgproj<<<dim3(784, 2), 256, 0, stream>>>(in1, in2, WimgB, b_img, ln_g, ln_b,
                                              emb_w, emb_h, outp, Xb0, Xb1);

  // 2) encoder: 2 layers x 2 images (sequential to share one TPG buffer)
  for (int l = 0; l < 2; ++l) {
    for (int img = 0; img < 2; ++img) {
      bf16* Xb = img ? Xb1 : Xb0;
      k_encgemm<<<784, 256, 0, stream>>>(Xb, WcatB + (size_t)l * NCAT * DM,
                                         enc_bt + l * 256, enc_bp + l * 256,
                                         enc_bg + l * 256, TPG);
      k_encattn<<<784, 256, 0, stream>>>(TPG, WbackB + (size_t)l * DM * CO,
                                         enc_bbk + l * DM,
                                         outp + (size_t)img * OUTSZ, Xb);
    }
  }

  // 3) cross attention (layer 1 conv weights), ctx -> d_out slots 2/3
  k_cross<<<dim3(13, 64, 16), 256, 0, stream>>>(Xb0, Xb1,
                                                ca_Wq + 512 * 9, ca_Wk + 512 * 9,
                                                ca_Wv + 512 * 9, outp);

  // 4) ctx + residual + LayerNorm in place (layer 1 LN params)
  k_ctxln<<<dim3(784, 2), 256, 0, stream>>>(outp, ca_ln_g + 512, ca_ln_b + 512);
}
